// MoeConv_34746285425195
// MI455X (gfx1250) — matrix operand order repaired and checked by reading, not compiled, not run on hardware
//
#include <hip/hip_runtime.h>

typedef __attribute__((ext_vector_type(16))) __bf16 v16bf;
typedef __attribute__((ext_vector_type(8)))  float  v8f;
typedef __attribute__((ext_vector_type(4)))  int    v4i_;

#define B_    8
#define CIN   32
#define COUT  32
#define H_    256
#define W_    256
#define E_    8
#define NKT   9      // K chunks of 32 (one per 3x3 tap)
#define NMT   18     // M tiles of 16: 16 expert tiles (8 experts x 2) + 2 shared
#define TILE_W 128
#define LCOLS  130   // TILE_W + 2 halo
#define ROWS   4     // output rows per block (amortizes weight staging)

#define WELEMS (NMT * NKT * 32 * 16)   // 82,944 bf16 = 165,888 B

#if defined(__AMDGCN__) && \
    __has_builtin(__builtin_amdgcn_global_load_async_to_lds_b128) && \
    __has_builtin(__builtin_amdgcn_s_wait_asynccnt)
#define HAVE_ASYNC_LDS 1
typedef __attribute__((address_space(1))) v4i_* as1_v4i;
typedef __attribute__((address_space(3))) v4i_* as3_v4i;
#endif

// ---------------------------------------------------------------------------
// Kernel 1: pre-pack [288 x 288] bf16 weight matrix directly into the
// WMMA A-fragment layout with contiguous K order (k = 16*(lane/16) + j).
// wpack[mt][kt][lane][j] : lane loads its v16bf with two b128 loads.
// ---------------------------------------------------------------------------
__global__ void pack_weights_kernel(const float* __restrict__ expert_w,
                                    const float* __restrict__ shared_w,
                                    __bf16* __restrict__ wpack) {
  int t = blockIdx.x * blockDim.x + threadIdx.x;      // NMT*NKT*32 = 5184
  if (t >= NMT * NKT * 32) return;
  int lane = t & 31;
  int kt   = (t >> 5) % NKT;
  int mt   = (t >> 5) / NKT;
  int kh = kt / 3, kw = kt % 3;
  int mrow  = mt * 16 + (lane & 15);
  int kbase = (lane < 16) ? 0 : 16;

  __bf16* dst = wpack + ((size_t)(mt * NKT + kt) * 32 + lane) * 16;
#pragma unroll
  for (int j = 0; j < 16; ++j) {
    int v = j >> 1, sub = j & 1;
    // A-fragment K mapping: element j -> K j (+16 hi lanes), contiguous to match B
    int klocal = kbase + (2 * v + sub);
    int cin = klocal;
    float wv;
    if (mrow < E_ * COUT) {
      int e = mrow >> 5, c = mrow & 31;
      wv = expert_w[(((size_t)(e * COUT + c) * CIN + cin) * 3 + kh) * 3 + kw];
    } else {
      int c = mrow - E_ * COUT;
      wv = shared_w[(((size_t)c * CIN + cin) * 3 + kh) * 3 + kw];
    }
    dst[j] = (__bf16)wv;
  }
}

// ---------------------------------------------------------------------------
// Kernel 2: gate conv (E=8) -> sigmoid -> top-2 on biased scores ->
// softmax of raw scores -> dense routing map [B,E,H,W] (f32) in workspace.
// ---------------------------------------------------------------------------
__global__ __launch_bounds__(256)
void gate_kernel(const float* __restrict__ x,
                 const float* __restrict__ gate_w,
                 const float* __restrict__ gate_bias,
                 float* __restrict__ dense) {
  __shared__ float gw[E_ * CIN * 9];
  for (int i = threadIdx.x; i < E_ * CIN * 9; i += blockDim.x) gw[i] = gate_w[i];
  __syncthreads();

  int p = blockIdx.x * blockDim.x + threadIdx.x;       // one pixel per thread
  int w = p % W_;
  int h = (p / W_) % H_;
  int b = p / (W_ * H_);

  float s[E_];
#pragma unroll
  for (int e = 0; e < E_; ++e) s[e] = 0.f;

  for (int cin = 0; cin < CIN; ++cin) {
    const float* xp = x + ((size_t)(b * CIN + cin)) * H_ * W_;
#pragma unroll
    for (int kh = 0; kh < 3; ++kh) {
      int hh = h + kh - 1;
      if (hh < 0 || hh >= H_) continue;
#pragma unroll
      for (int kw = 0; kw < 3; ++kw) {
        int ww = w + kw - 1;
        if (ww < 0 || ww >= W_) continue;
        float xv = xp[hh * W_ + ww];
        const float* g = &gw[cin * 9 + kh * 3 + kw];
#pragma unroll
        for (int e = 0; e < E_; ++e) s[e] += xv * g[e * CIN * 9];
      }
    }
  }

  float sc[E_], bi[E_];
#pragma unroll
  for (int e = 0; e < E_; ++e) {
    sc[e] = 1.f / (1.f + __expf(-s[e]));
    bi[e] = sc[e] + gate_bias[e];
  }
  // top-2 on biased scores, first-occurrence tie-break like lax.top_k
  int i0 = 0;
#pragma unroll
  for (int e = 1; e < E_; ++e) if (bi[e] > bi[i0]) i0 = e;
  int i1 = -1;
  float b1 = -1e30f;
#pragma unroll
  for (int e = 0; e < E_; ++e) {
    if (e == i0) continue;
    if (bi[e] > b1) { b1 = bi[e]; i1 = e; }
  }
  float s0 = sc[i0], s1 = sc[i1];
  float m  = fmaxf(s0, s1);
  float e0 = __expf(s0 - m), e1 = __expf(s1 - m);
  float inv = 1.f / (e0 + e1);

  float outw[E_];
#pragma unroll
  for (int e = 0; e < E_; ++e) outw[e] = 0.f;
  outw[i0] = e0 * inv;   // ROUTE_SCALE = 1
  outw[i1] = e1 * inv;

  size_t base = ((size_t)b * E_ * H_ + h) * W_ + w;
#pragma unroll
  for (int e = 0; e < E_; ++e) dense[base + (size_t)e * H_ * W_] = outw[e];
}

// ---------------------------------------------------------------------------
// Kernel 3: implicit-GEMM expert+shared conv with bf16 WMMA, routed combine.
// Block = 256 threads (8 waves). Block covers ROWS rows x 128 pixels.
// All packed weights (162 KB) staged into LDS once per block (async path),
// x halo tile staged per row. A-fragments come from ds_load_b128, double-
// buffered across K-steps; each expert runs two independent WMMA chains.
// ---------------------------------------------------------------------------
union AF { v16bf v; uint4 q[2]; };

__global__ __launch_bounds__(256)
void expert_kernel(const float* __restrict__ x,
                   const float* __restrict__ dense,
                   const float* __restrict__ expert_b,
                   const float* __restrict__ shared_b,
                   const __bf16* __restrict__ wpack,
                   float* __restrict__ out) {
  __shared__ __bf16 wlds[WELEMS];          // 165,888 B packed weights
  __shared__ __bf16 xt[3 * LCOLS * CIN];   //  24,960 B x halo tile

  int tid = threadIdx.x;
  int blk = blockIdx.x;                    // B * (H/ROWS) * (W/TILE_W) = 1024
  int wt  = (blk & 1) * TILE_W;
  int rest = blk >> 1;
  int h0 = (rest % (H_ / ROWS)) * ROWS;
  int b  = rest / (H_ / ROWS);

  // ---- stage all packed weights into LDS (CDNA5 async copy if available) ----
  {
    const int NQ = WELEMS * 2 / 16;        // 10,368 x 16B
    const uint4* src = (const uint4*)wpack;
    uint4* dst = (uint4*)wlds;
#ifdef HAVE_ASYNC_LDS
    for (int i = tid; i < NQ; i += 256) {
      __builtin_amdgcn_global_load_async_to_lds_b128(
          (as1_v4i)(src + i), (as3_v4i)(dst + i),
          /*offset=*/0, /*cpol=*/0);
    }
    __builtin_amdgcn_s_wait_asynccnt(0);
#else
    for (int i = tid; i < NQ; i += 256) dst[i] = src[i];
#endif
  }

  int lane = tid & 31;
  int wv   = tid >> 5;        // wave id 0..7 -> 16-pixel subtile
  int n    = lane & 15;       // N column (pixel) within tile
  int hi   = lane >> 4;       // lane-half selector
  int px_local = wv * 16 + n; // 0..127

  // pre-load bias fragments (same for every row)
  float eb[E_][2][8];
#pragma unroll
  for (int e = 0; e < E_; ++e)
#pragma unroll
    for (int half = 0; half < 2; ++half)
#pragma unroll
      for (int i = 0; i < 8; ++i)
        eb[e][half][i] = expert_b[e * COUT + half * 16 + hi * 8 + i];
  float sb[2][8];
#pragma unroll
  for (int half = 0; half < 2; ++half)
#pragma unroll
    for (int i = 0; i < 8; ++i)
      sb[half][i] = shared_b[half * 16 + hi * 8 + i];

#pragma unroll 1
  for (int row = 0; row < ROWS; ++row) {
    int h = h0 + row;
    __syncthreads();   // weights ready / previous-row readers done

    // ---- stage x halo tile (f32 -> bf16), coalesced along cols ----
    const int NLOAD = 3 * LCOLS * CIN;     // 12,480 elements
    for (int i = tid; i < NLOAD; i += 256) {
      int col = i % LCOLS;
      int rc  = i / LCOLS;
      int cin = rc & 31;
      int r   = rc >> 5;
      int hh = h + r - 1;
      int ww = wt + col - 1;
      float v = 0.f;
      if (hh >= 0 && hh < H_ && ww >= 0 && ww < W_)
        v = x[((size_t)(b * CIN + cin) * H_ + hh) * W_ + ww];
      xt[(r * LCOLS + col) * CIN + cin] = (__bf16)v;
    }
    __syncthreads();

    // ---- build the 9 B fragments (K=32 x N=16) from LDS ----
    v16bf bfrag[NKT];
#pragma unroll
    for (int kt = 0; kt < NKT; ++kt) {
      int kh = kt / 3, kw = kt % 3;
      int col = px_local + kw;
      const __bf16* src = &xt[(kh * LCOLS + col) * CIN + hi * 16];
      AF u;
      u.q[0] = *(const uint4*)(src);
      u.q[1] = *(const uint4*)(src + 8);
      bfrag[kt] = u.v;
    }

    v8f out0 = {0.f, 0.f, 0.f, 0.f, 0.f, 0.f, 0.f, 0.f};
    v8f out1 = {0.f, 0.f, 0.f, 0.f, 0.f, 0.f, 0.f, 0.f};

    const float* dense_p =
        dense + ((size_t)b * E_ * H_ + h) * W_ + wt + wv * 16;

    // ---- experts: two independent WMMA chains per expert ----
#pragma unroll 1
    for (int e = 0; e < E_; ++e) {
      float dw = dense_p[(size_t)e * H_ * W_ + n];
      v8f acc0, acc1;
#pragma unroll
      for (int i = 0; i < 8; ++i) { acc0[i] = eb[e][0][i]; acc1[i] = eb[e][1][i]; }

      AF a0, a1, p0, p1;
      {
        const uint4* q0 = (const uint4*)(wlds + ((2 * e * NKT) * 32 + lane) * 16);
        const uint4* q1 = (const uint4*)(wlds + (((2 * e + 1) * NKT) * 32 + lane) * 16);
        a0.q[0] = q0[0]; a0.q[1] = q0[1];
        a1.q[0] = q1[0]; a1.q[1] = q1[1];
      }
#pragma unroll
      for (int kt = 0; kt < NKT; ++kt) {
        if (kt < NKT - 1) {   // prefetch next K-step's A fragments under the WMMAs
          const uint4* q0 =
              (const uint4*)(wlds + ((2 * e * NKT + kt + 1) * 32 + lane) * 16);
          const uint4* q1 =
              (const uint4*)(wlds + (((2 * e + 1) * NKT + kt + 1) * 32 + lane) * 16);
          p0.q[0] = q0[0]; p0.q[1] = q0[1];
          p1.q[0] = q1[0]; p1.q[1] = q1[1];
        }
        acc0 = __builtin_amdgcn_wmma_f32_16x16x32_bf16(
            false, a0.v, false, bfrag[kt], (short)0, acc0, false, false);
        acc1 = __builtin_amdgcn_wmma_f32_16x16x32_bf16(
            false, a1.v, false, bfrag[kt], (short)0, acc1, false, false);
        if (kt < NKT - 1) { a0 = p0; a1 = p1; }
      }
#pragma unroll
      for (int i = 0; i < 8; ++i) {
        out0[i] += dw * acc0[i];
        out1[i] += dw * acc1[i];
      }
    }

    // ---- shared expert (unweighted), tiles mt=16,17 ----
    {
      v8f acc0, acc1;
#pragma unroll
      for (int i = 0; i < 8; ++i) { acc0[i] = sb[0][i]; acc1[i] = sb[1][i]; }
#pragma unroll
      for (int kt = 0; kt < NKT; ++kt) {
        AF a0, a1;
        const uint4* q0 = (const uint4*)(wlds + ((16 * NKT + kt) * 32 + lane) * 16);
        const uint4* q1 = (const uint4*)(wlds + ((17 * NKT + kt) * 32 + lane) * 16);
        a0.q[0] = q0[0]; a0.q[1] = q0[1];
        a1.q[0] = q1[0]; a1.q[1] = q1[1];
        acc0 = __builtin_amdgcn_wmma_f32_16x16x32_bf16(
            false, a0.v, false, bfrag[kt], (short)0, acc0, false, false);
        acc1 = __builtin_amdgcn_wmma_f32_16x16x32_bf16(
            false, a1.v, false, bfrag[kt], (short)0, acc1, false, false);
      }
#pragma unroll
      for (int i = 0; i < 8; ++i) { out0[i] += acc0[i]; out1[i] += acc1[i]; }
    }

    // ---- store: channel (i + 8*hi) and (16 + i + 8*hi), pixel wt+wv*16+n ----
    float* op = out + ((size_t)b * COUT * H_ + h) * W_ + wt + wv * 16 + n;
#pragma unroll
    for (int i = 0; i < 8; ++i) {
      int c0 = i + 8 * hi;
      op[(size_t)c0 * H_ * W_]        = out0[i];
      op[(size_t)(16 + c0) * H_ * W_] = out1[i];
    }
  }
}

// ---------------------------------------------------------------------------
extern "C" void kernel_launch(void* const* d_in, const int* in_sizes, int n_in,
                              void* d_out, int out_size, void* d_ws, size_t ws_size,
                              hipStream_t stream) {
  const float* x        = (const float*)d_in[0];
  const float* gate_w   = (const float*)d_in[1];
  const float* gate_bias= (const float*)d_in[2];
  const float* expert_w = (const float*)d_in[3];
  const float* expert_b = (const float*)d_in[4];
  const float* shared_w = (const float*)d_in[5];
  const float* shared_b = (const float*)d_in[6];
  float* out = (float*)d_out;

  float*  dense = (float*)d_ws;                                       // 16 MB
  __bf16* wpack = (__bf16*)((char*)d_ws +
                            (size_t)B_ * E_ * H_ * W_ * sizeof(float)); // 162 KB

  pack_weights_kernel<<<(NMT * NKT * 32 + 255) / 256, 256, 0, stream>>>(
      expert_w, shared_w, wpack);

  gate_kernel<<<(B_ * H_ * W_) / 256, 256, 0, stream>>>(
      x, gate_w, gate_bias, dense);

  expert_kernel<<<B_ * (H_ / ROWS) * (W_ / TILE_W), 256, 0, stream>>>(
      x, dense, expert_b, shared_b, wpack, out);
}
